// COMPOLBlock2d_37091337568326
// MI455X (gfx1250) — compile-verified, matrix operand order checked by reading, not run on hardware
//
#include <hip/hip_runtime.h>
#include <hip/hip_bf16.h>

// ---------------- problem constants ----------------
constexpr int B_ = 8, C_ = 64, H_ = 128, W_ = 128, M_ = 4;
constexpr int HW = H_ * W_;               // 16384
constexpr int NPTS = B_ * HW;             // 131072 spatial points
constexpr float C2PI = 6.283185307179586f / 128.f;

// workspace layout (float offsets)
constexpr size_t W1FO = 0;          // f16 A-frags: [mbc=2048][kt=8][512h]  = 4,194,304 fl
constexpr size_t ZO   = 4194304;    // f32 [m][ky][kxi=32][ri][b=8][c=64]  = 2,097,152 fl
constexpr size_t OO   = 6291456;    // f32, same shape as Z                 = 2,097,152 fl
constexpr size_t WIFO = 8388608;    // f16 A-frags: [mbc*8+ht=16384][512h]  = 4,194,304 fl
constexpr size_t SPO  = 12582912;   // f32 [m][b][c][h][w]                  = 33,554,432 fl
constexpr size_t FRO  = 46137344;   // f16 B-fragment pool (176 tiles)      =    45,056 fl

// B-fragment pool tile ids (each tile = 32 lanes x 16 halves = 512 halves)
constexpr int T_ATN_C = 0,  T_ATN_Q = 8,  T_ATN_K = 16, T_ATN_A = 24;
constexpr int T_OUT_W = 32, T_OUT_Z = 64;
constexpr int T_DFTW  = 96, T_DFTH = 104, T_IDFTH = 136, T_IDFTW = 168;
constexpr int N_TILES = 176;

// ---------------- WMMA helpers (CDNA5 wave32) ----------------
typedef __attribute__((ext_vector_type(16))) _Float16 v16h;
typedef __attribute__((ext_vector_type(8)))  float    v8f;

__device__ inline int lane_id() { return threadIdx.x & 31; }

// A tile 16x32 f16 assembled from arbitrary source (transpose gather)
template <class F>
__device__ inline v16h mk_a(F f) {
  const int L = lane_id(), row = L & 15, half = L >> 4;
  v16h a;
#pragma unroll
  for (int e = 0; e < 16; ++e) {
    int k = ((e < 8) ? 0 : 16) + 8 * half + (e & 7);
    a[e] = (_Float16)f(row, k);
  }
  return a;
}
// B tile 32x16 f16 assembled element-wise (only used for single-use weights)
template <class F>
__device__ inline v16h mk_b(F f) {
  const int L = lane_id(), col = L & 15, half = L >> 4;
  v16h b;
#pragma unroll
  for (int e = 0; e < 16; ++e) {
    int k = ((e < 8) ? 0 : 16) + 8 * half + (e & 7);
    b[e] = (_Float16)f(k, col);
  }
  return b;
}
// load a pre-swizzled fragment: one aligned 32-byte vector load per lane
__device__ inline v16h ld_frag(const _Float16* frag, int t) {
  return *(const v16h*)(frag + (size_t)t * 512 + lane_id() * 16);
}
// C/D tile 16x16 f32: VGPR r, lane L -> row = r + 8*(L>>4), col = L&15
template <class F>
__device__ inline void st_c(const v8f& d, F f) {
  const int L = lane_id(), col = L & 15, half = L >> 4;
#pragma unroll
  for (int r = 0; r < 8; ++r) f(r + 8 * half, col, d[r]);
}
// position of element (row, k) inside an A fragment (halves)
__device__ inline int afrag_pos(int row, int k32) {
  int hl = (k32 >> 3) & 1;
  int e  = (k32 & 7) + ((k32 & 16) ? 8 : 0);
  return (row + 16 * hl) * 16 + e;
}
__device__ inline v8f wmma(const v16h& a, const v16h& b, const v8f& c) {
  return __builtin_amdgcn_wmma_f32_16x16x32_f16(false, a, false, b, (short)0, c,
                                                false, false);
}
__device__ inline v8f zero8() { v8f z;
#pragma unroll
  for (int r = 0; r < 8; ++r) z[r] = 0.f; return z; }

// ============================================================
// Kernel 0: pack all reused B panels (weights + DFT twiddles) as f16 fragments
// ============================================================
__global__ __launch_bounds__(256) void k_pack(
    const float* __restrict__ c_w, const float* __restrict__ wq_w,
    const float* __restrict__ wk_w, const float* __restrict__ wa_w,
    const float* __restrict__ w_w, const float* __restrict__ wz_w,
    float* __restrict__ ws) {
  const int t = blockIdx.x * 8 + (threadIdx.x >> 5);   // 176 tiles, grid=22
  const int L = lane_id(), col = L & 15, half = L >> 4;
  v16h b;
#pragma unroll
  for (int e = 0; e < 16; ++e) {
    int k = ((e < 8) ? 0 : 16) + 8 * half + (e & 7);
    float v;
    if (t < T_ATN_Q)       { int i=t-T_ATN_C; int kt=i>>2, nt=i&3; v = c_w [(nt*16+col)*64 + kt*32 + k]; }
    else if (t < T_ATN_K)  { int i=t-T_ATN_Q; int kt=i>>2, nt=i&3; v = wq_w[(nt*16+col)*64 + kt*32 + k]; }
    else if (t < T_ATN_A)  { int i=t-T_ATN_K; int kt=i>>2, nt=i&3; v = wk_w[(nt*16+col)*64 + kt*32 + k]; }
    else if (t < T_OUT_W)  { int i=t-T_ATN_A; int kt=i>>2, nt=i&3; v = wa_w[(nt*16+col)*64 + kt*32 + k]; }
    else if (t < T_OUT_Z)  { int i=t-T_OUT_W; int m=i>>3, kt=(i>>2)&1, nt=i&3;
                             v = w_w [(m*64+nt*16+col)*64 + kt*32 + k]; }
    else if (t < T_DFTW)   { int i=t-T_OUT_Z; int m=i>>3, kt=(i>>2)&1, nt=i&3;
                             v = wz_w[(m*64+nt*16+col)*64 + kt*32 + k]; }
    else if (t < T_DFTH)   { int i=t-T_DFTW; int kt=i>>1, ri=i&1; int w=kt*32+k;
                             float th = C2PI * (float)((col*w)&127);
                             v = ri ? -__sinf(th) : __cosf(th); }
    else if (t < T_IDFTH)  { int i=t-T_DFTH; int kt=i>>2, ntk=(i>>1)&1, ri=i&1;
                             int kk=kt*32+k, hh=kk&127, im=kk>>7;
                             int kxi=ntk*16+col, kx = kxi<16 ? kxi : 96+kxi;
                             float th = C2PI * (float)((kx*hh)&127);
                             v = (ri==0) ? (im ? __sinf(th) :  __cosf(th))
                                         : (im ? __cosf(th) : -__sinf(th)); }
    else if (t < T_IDFTW)  { int i=t-T_IDFTH; int kt=i>>4, nt=i&15, ro=nt>>3, h0=(nt&7)*16;
                             int kk=kt*32+k, kxi=kk&31, im=kk>>5, kx = kxi<16 ? kxi : 96+kxi;
                             int h=h0+col;
                             float th = C2PI * (float)((kx*h)&127);
                             v = (ro==0) ? (im ? -__sinf(th) : __cosf(th))
                                         : (im ?  __cosf(th) : __sinf(th)); }
    else                   { int nt=t-T_IDFTW; int ky=k&15, im=k>>4, w=nt*16+col;
                             float s = (ky==0 ? 1.f : 2.f) * (1.f/16384.f);
                             float th = C2PI * (float)((ky*w)&127);
                             v = im ? -s*__sinf(th) : s*__cosf(th); }
    b[e] = (_Float16)v;
  }
  _Float16* frag = (_Float16*)(ws + FRO);
  *(v16h*)(frag + (size_t)t * 512 + L * 16) = b;
}

// ============================================================
// Kernel 1: ATN aggregation -> z  (one wave per 16 pixels)
// ============================================================
__global__ __launch_bounds__(32) void k_atn(
    const float* __restrict__ x0, const float* __restrict__ x1,
    const float* __restrict__ x2, const float* __restrict__ x3,
    const float* __restrict__ c_b, const float* __restrict__ wq_b,
    const float* __restrict__ wk_b, const float* __restrict__ wa_b,
    const float* __restrict__ ws, float* __restrict__ z_out) {
  __shared__ alignas(32) _Float16 sVf[2][512];   // staged Vc in A-fragment layout
  __shared__ float sQ[16][64];
  __shared__ float sK[4][16][64];
  __shared__ float sA[4][16][64];
  const _Float16* frag = (const _Float16*)(ws + FRO);
  const int L = lane_id();
  const int p0 = blockIdx.x * 16;
  const int b  = p0 / HW;
  const int hw = p0 % HW;
  const float* xs[4] = {x0, x1, x2, x3};

  v8f vsum[4];
#pragma unroll
  for (int nt = 0; nt < 4; ++nt) vsum[nt] = zero8();

  for (int m = 0; m < 4; ++m) {
    const float* xm = xs[m];
    v8f acc[4];
#pragma unroll
    for (int nt = 0; nt < 4; ++nt) {
      float bv = c_b[nt * 16 + (L & 15)];
      v8f t;
#pragma unroll
      for (int r = 0; r < 8; ++r) t[r] = bv;
      acc[nt] = t;
    }
#pragma unroll
    for (int kt = 0; kt < 2; ++kt) {
      v16h a = mk_a([&](int r, int k) -> float {
        return xm[(b * 64 + kt * 32 + k) * HW + hw + r];
      });
#pragma unroll
      for (int nt = 0; nt < 4; ++nt)
        acc[nt] = wmma(a, ld_frag(frag, T_ATN_C + kt * 4 + nt), acc[nt]);
    }
#pragma unroll
    for (int nt = 0; nt < 4; ++nt) vsum[nt] += acc[nt];
    __syncthreads();
#pragma unroll
    for (int nt = 0; nt < 4; ++nt)
      st_c(acc[nt], [&](int r, int n, float v) {
        int ch = nt * 16 + n;
        sVf[ch >> 5][afrag_pos(r, ch & 31)] = (_Float16)v;
      });
    __syncthreads();
    v8f aK[4], aA[4];
#pragma unroll
    for (int nt = 0; nt < 4; ++nt) {
      float bk = wk_b[nt * 16 + (L & 15)], ba = wa_b[nt * 16 + (L & 15)];
      v8f tk, ta;
#pragma unroll
      for (int r = 0; r < 8; ++r) { tk[r] = bk; ta[r] = ba; }
      aK[nt] = tk; aA[nt] = ta;
    }
#pragma unroll
    for (int kt = 0; kt < 2; ++kt) {
      v16h a = *(const v16h*)(&sVf[kt][0] + L * 16);
#pragma unroll
      for (int nt = 0; nt < 4; ++nt) {
        aK[nt] = wmma(a, ld_frag(frag, T_ATN_K + kt * 4 + nt), aK[nt]);
        aA[nt] = wmma(a, ld_frag(frag, T_ATN_A + kt * 4 + nt), aA[nt]);
      }
    }
#pragma unroll
    for (int nt = 0; nt < 4; ++nt) {
      st_c(aK[nt], [&](int r, int n, float v) { sK[m][r][nt * 16 + n] = v; });
      st_c(aA[nt], [&](int r, int n, float v) { sA[m][r][nt * 16 + n] = v; });
    }
  }
  // Q = mean(Vc) @ wq^T + b
  __syncthreads();
#pragma unroll
  for (int nt = 0; nt < 4; ++nt)
    st_c(vsum[nt], [&](int r, int n, float v) {
      int ch = nt * 16 + n;
      sVf[ch >> 5][afrag_pos(r, ch & 31)] = (_Float16)(v * 0.25f);
    });
  __syncthreads();
  v8f q[4];
#pragma unroll
  for (int nt = 0; nt < 4; ++nt) {
    float bq = wq_b[nt * 16 + (L & 15)];
    v8f t;
#pragma unroll
    for (int r = 0; r < 8; ++r) t[r] = bq;
    q[nt] = t;
  }
#pragma unroll
  for (int kt = 0; kt < 2; ++kt) {
    v16h a = *(const v16h*)(&sVf[kt][0] + L * 16);
#pragma unroll
    for (int nt = 0; nt < 4; ++nt)
      q[nt] = wmma(a, ld_frag(frag, T_ATN_Q + kt * 4 + nt), q[nt]);
  }
#pragma unroll
  for (int nt = 0; nt < 4; ++nt)
    st_c(q[nt], [&](int r, int n, float v) { sQ[r][nt * 16 + n] = v; });
  __syncthreads();

  // per-pixel scores / softmax over M=4 / z  (lanes split channel halves)
  const int p = L & 15, ch0 = (L >> 4) * 32;
  float sc[4];
#pragma unroll
  for (int m = 0; m < 4; ++m) {
    float d = 0.f;
    for (int c = 0; c < 64; ++c) d += sQ[p][c] * sK[m][p][c];
    sc[m] = d * 0.125f;  // 1/sqrt(64)
  }
  float mx = fmaxf(fmaxf(sc[0], sc[1]), fmaxf(sc[2], sc[3]));
  float e[4], s = 0.f;
#pragma unroll
  for (int m = 0; m < 4; ++m) { e[m] = __expf(sc[m] - mx); s += e[m]; }
  float inv = 1.f / s;
  for (int c = ch0; c < ch0 + 32; ++c) {
    float zv = 0.f;
#pragma unroll
    for (int m = 0; m < 4; ++m) zv += e[m] * inv * sA[m][p][c];
    z_out[(b * 64 + c) * HW + hw + p] = zv;
  }
}

// ============================================================
// Kernel 2: forward partial DFT along W (16 ky modes); writes f16 A-fragments
// ============================================================
__global__ __launch_bounds__(256) void k_dft_w(
    const float* __restrict__ x0, const float* __restrict__ x1,
    const float* __restrict__ x2, const float* __restrict__ x3,
    float* __restrict__ ws) {
  const _Float16* frag = (const _Float16*)(ws + FRO);
  const int wv  = blockIdx.x * 8 + (threadIdx.x >> 5);   // 16384 row-tiles
  const int mbc = wv >> 3;
  const int h0  = (wv & 7) * 16;
  const int m = mbc >> 9, bc = mbc & 511;
  const float* xs[4] = {x0, x1, x2, x3};
  const float* xm = xs[m] + (size_t)bc * HW;
  v8f acc[2]; acc[0] = zero8(); acc[1] = zero8();
#pragma unroll
  for (int kt = 0; kt < 4; ++kt) {
    v16h a = mk_a([&](int r, int k) -> float { return xm[(h0 + r) * 128 + kt * 32 + k]; });
    acc[0] = wmma(a, ld_frag(frag, T_DFTW + kt * 2 + 0), acc[0]);
    acc[1] = wmma(a, ld_frag(frag, T_DFTW + kt * 2 + 1), acc[1]);
  }
  // store transposed into A-fragment layout for k_dft_h: row=ky, kk=ri*128+h
  _Float16* W1 = (_Float16*)(ws + W1FO) + (size_t)mbc * 4096;
#pragma unroll
  for (int ri = 0; ri < 2; ++ri)
    st_c(acc[ri], [&](int r, int n, float v) {
      int kk = ri * 128 + h0 + r;
      W1[(size_t)(kk >> 5) * 512 + afrag_pos(n, kk & 31)] = (_Float16)v;
    });
}

// ============================================================
// Kernel 3: forward DFT along H, 32 kept kx modes; complex via concat-K=256
// ============================================================
__global__ __launch_bounds__(256) void k_dft_h(float* __restrict__ ws) {
  const _Float16* frag = (const _Float16*)(ws + FRO);
  const int mbc = blockIdx.x * 8 + (threadIdx.x >> 5);   // 2048
  const int m = mbc >> 9, bc = mbc & 511;
  const int L = lane_id();
  const _Float16* W1 = (const _Float16*)(ws + W1FO) + (size_t)mbc * 4096;
  v8f acc[2][2];
#pragma unroll
  for (int i = 0; i < 2; ++i) { acc[i][0] = zero8(); acc[i][1] = zero8(); }
#pragma unroll
  for (int kt = 0; kt < 8; ++kt) {
    v16h a = *(const v16h*)(W1 + (size_t)kt * 512 + L * 16);
#pragma unroll
    for (int ntk = 0; ntk < 2; ++ntk) {
      acc[0][ntk] = wmma(a, ld_frag(frag, T_DFTH + kt * 4 + ntk * 2 + 0), acc[0][ntk]);
      acc[1][ntk] = wmma(a, ld_frag(frag, T_DFTH + kt * 4 + ntk * 2 + 1), acc[1][ntk]);
    }
  }
  float* Z = ws + ZO;
#pragma unroll
  for (int ri = 0; ri < 2; ++ri)
#pragma unroll
    for (int ntk = 0; ntk < 2; ++ntk)
      st_c(acc[ri][ntk], [&](int r, int n, float v) {
        int ky = r, kxi = ntk * 16 + n;
        Z[((size_t)((m * 16 + ky) * 32 + kxi) * 2 + ri) * 512 + bc] = v;
      });
}

// ============================================================
// Kernel 4: spectral channel mix, complex [8xC]x[CxC] per (m,kxi,ky)
// (weights used exactly once -> inline gather, no packing)
// ============================================================
__global__ __launch_bounds__(256) void k_mix(
    const float* __restrict__ sw1re, const float* __restrict__ sw1im,
    const float* __restrict__ sw2re, const float* __restrict__ sw2im,
    float* __restrict__ ws) {
  const int u = blockIdx.x * 8 + (threadIdx.x >> 5);   // 2048 = m*512 + kxi*16 + ky
  const int m = u >> 9, kxi = (u >> 4) & 31, ky = u & 15;
  const float *Wre, *Wim; int kxp;
  if (kxi < 16) { Wre = sw1re; Wim = sw1im; kxp = kxi; }
  else          { Wre = sw2re; Wim = sw2im; kxp = kxi - 16; }
  const float* Zb = ws + ZO + ((size_t)((m * 16 + ky) * 32 + kxi) * 2) * 512;
  v8f acc[2][4];
#pragma unroll
  for (int i = 0; i < 2; ++i)
#pragma unroll
    for (int j = 0; j < 4; ++j) acc[i][j] = zero8();
#pragma unroll
  for (int kt = 0; kt < 4; ++kt) {
    v16h a = mk_a([&](int r, int k) -> float {
      if (r >= 8) return 0.f;                      // only B_=8 rows valid
      int kk = kt * 32 + k;                        // concat-K: re(0..63) | im(64..127)
      return Zb[(size_t)(kk >> 6) * 512 + r * 64 + (kk & 63)];
    });
#pragma unroll
    for (int nt = 0; nt < 4; ++nt) {
      v16h br = mk_b([&](int k, int n) -> float {
        int kk = kt * 32 + k, i = kk & 63, o = nt * 16 + n;
        size_t wi = ((size_t)((m * 64 + i) * 64 + o) * 16 + kxp) * 16 + ky;
        return (kk < 64) ? Wre[wi] : -Wim[wi];
      });
      acc[0][nt] = wmma(a, br, acc[0][nt]);
      v16h bi = mk_b([&](int k, int n) -> float {
        int kk = kt * 32 + k, i = kk & 63, o = nt * 16 + n;
        size_t wi = ((size_t)((m * 64 + i) * 64 + o) * 16 + kxp) * 16 + ky;
        return (kk < 64) ? Wim[wi] : Wre[wi];
      });
      acc[1][nt] = wmma(a, bi, acc[1][nt]);
    }
  }
  float* Ob = ws + OO + ((size_t)((m * 16 + ky) * 32 + kxi) * 2) * 512;
#pragma unroll
  for (int ri = 0; ri < 2; ++ri)
#pragma unroll
    for (int nt = 0; nt < 4; ++nt)
      st_c(acc[ri][nt], [&](int r, int n, float v) {
        if (r < 8) Ob[(size_t)ri * 512 + r * 64 + nt * 16 + n] = v;
      });
}

// ============================================================
// Kernel 5: inverse DFT along H (32 modes -> 128 h); writes f16 A-fragments
// ============================================================
__global__ __launch_bounds__(256) void k_idft_h(float* __restrict__ ws) {
  const _Float16* frag = (const _Float16*)(ws + FRO);
  const int u   = blockIdx.x * 8 + (threadIdx.x >> 5);  // 2048 = mky*32 + bc-tile
  const int mky = u >> 5;
  const int bc0 = (u & 31) * 16;
  const int m = mky >> 4, ky = mky & 15;
  const float* Ob = ws + OO + (size_t)mky * 32 * 2 * 512;
  v8f acc[16];
#pragma unroll
  for (int i = 0; i < 16; ++i) acc[i] = zero8();
#pragma unroll
  for (int kt = 0; kt < 2; ++kt) {
    v16h a = mk_a([&](int r, int k) -> float {
      int kk = kt * 32 + k, kxi = kk & 31, ri = kk >> 5;
      return Ob[((size_t)kxi * 2 + ri) * 512 + bc0 + r];
    });
#pragma unroll
    for (int nt = 0; nt < 16; ++nt)
      acc[nt] = wmma(a, ld_frag(frag, T_IDFTH + kt * 16 + nt), acc[nt]);
  }
  // store transposed into A-fragment layout for k_idft_w: row=h&15, k=ro*16+ky
  _Float16* WIF = (_Float16*)(ws + WIFO);
#pragma unroll
  for (int nt = 0; nt < 16; ++nt) {
    const int ro = nt >> 3, h0 = (nt & 7) * 16;
    st_c(acc[nt], [&](int r, int n, float v) {
      int bc = bc0 + r, h = h0 + n, kk = ro * 16 + ky;
      size_t fid = (size_t)(m * 512 + bc) * 8 + (h >> 4);
      WIF[fid * 512 + afrag_pos(h & 15, kk)] = (_Float16)v;
    });
  }
}

// ============================================================
// Kernel 6: inverse rfft along W (16 modes -> 128 w, real), K=32, 1/(H*W) folded
// ============================================================
__global__ __launch_bounds__(256) void k_idft_w(float* __restrict__ ws) {
  const _Float16* frag = (const _Float16*)(ws + FRO);
  const int wv  = blockIdx.x * 8 + (threadIdx.x >> 5);  // 16384 row-tiles
  const int mbc = wv >> 3;
  const int h0  = (wv & 7) * 16;
  const int L = lane_id();
  const _Float16* WIF = (const _Float16*)(ws + WIFO);
  v8f acc[8];
#pragma unroll
  for (int i = 0; i < 8; ++i) acc[i] = zero8();
  {
    v16h a = *(const v16h*)(WIF + ((size_t)mbc * 8 + (wv & 7)) * 512 + L * 16);
#pragma unroll
    for (int nt = 0; nt < 8; ++nt)
      acc[nt] = wmma(a, ld_frag(frag, T_IDFTW + nt), acc[nt]);
  }
  float* SP = ws + SPO + (size_t)mbc * 16384;
#pragma unroll
  for (int nt = 0; nt < 8; ++nt)
    st_c(acc[nt], [&](int r, int n, float v) { SP[(h0 + r) * 128 + nt * 16 + n] = v; });
}

// ============================================================
// Kernel 7: out = gelu(sp + x@w^T + z@wz^T + biases)
// ============================================================
__global__ __launch_bounds__(256) void k_out(
    const float* __restrict__ x0, const float* __restrict__ x1,
    const float* __restrict__ x2, const float* __restrict__ x3,
    const float* __restrict__ w_b, const float* __restrict__ wz_b,
    const float* __restrict__ zbuf, const float* __restrict__ ws,
    float* __restrict__ out) {
  const _Float16* frag = (const _Float16*)(ws + FRO);
  const int wv   = blockIdx.x * 8 + (threadIdx.x >> 5);  // 32768 = m*8192 + tile
  const int m    = wv >> 13;
  const int tile = wv & 8191;
  const int p0 = tile * 16;
  const int b = p0 / HW, hw = p0 % HW;
  const int L = lane_id();
  const float* xs[4] = {x0, x1, x2, x3};
  const float* xm = xs[m];
  __builtin_prefetch(xm + (size_t)b * 64 * HW + hw, 0, 0);

  v8f acc[4];
#pragma unroll
  for (int nt = 0; nt < 4; ++nt) {
    float bv = w_b[m * 64 + nt * 16 + (L & 15)] + wz_b[m * 64 + nt * 16 + (L & 15)];
    v8f t;
#pragma unroll
    for (int r = 0; r < 8; ++r) t[r] = bv;
    acc[nt] = t;
  }
#pragma unroll
  for (int kt = 0; kt < 2; ++kt) {
    v16h a = mk_a([&](int r, int k) -> float {
      return xm[(b * 64 + kt * 32 + k) * HW + hw + r];
    });
#pragma unroll
    for (int nt = 0; nt < 4; ++nt)
      acc[nt] = wmma(a, ld_frag(frag, T_OUT_W + m * 8 + kt * 4 + nt), acc[nt]);
  }
#pragma unroll
  for (int kt = 0; kt < 2; ++kt) {
    v16h a = mk_a([&](int r, int k) -> float {
      return zbuf[(b * 64 + kt * 32 + k) * HW + hw + r];
    });
#pragma unroll
    for (int nt = 0; nt < 4; ++nt)
      acc[nt] = wmma(a, ld_frag(frag, T_OUT_Z + m * 8 + kt * 4 + nt), acc[nt]);
  }
  const float* SP = ws + SPO;
#pragma unroll
  for (int nt = 0; nt < 4; ++nt) {
    st_c(acc[nt], [&](int r, int n, float v) {
      int c = nt * 16 + n;
      float x = v + SP[(size_t)(m * 512 + b * 64 + c) * 16384 + hw + r];
      float g = 0.5f * x * (1.f + erff(x * 0.70710678118f));
      out[(size_t)((m * 8 + b) * 64 + c) * 16384 + hw + r] = g;
    });
  }
}

// ============================================================
extern "C" void kernel_launch(void* const* d_in, const int* in_sizes, int n_in,
                              void* d_out, int out_size, void* d_ws, size_t ws_size,
                              hipStream_t stream) {
  const float* x0 = (const float*)d_in[0];
  const float* x1 = (const float*)d_in[1];
  const float* x2 = (const float*)d_in[2];
  const float* x3 = (const float*)d_in[3];
  const float* c_w = (const float*)d_in[4];   const float* c_b = (const float*)d_in[5];
  const float* wq_w = (const float*)d_in[6];  const float* wq_b = (const float*)d_in[7];
  const float* wk_w = (const float*)d_in[8];  const float* wk_b = (const float*)d_in[9];
  const float* wa_w = (const float*)d_in[10]; const float* wa_b = (const float*)d_in[11];
  const float* w_w = (const float*)d_in[12];  const float* w_b = (const float*)d_in[13];
  const float* wz_w = (const float*)d_in[14]; const float* wz_b = (const float*)d_in[15];
  const float* sw1re = (const float*)d_in[16]; const float* sw1im = (const float*)d_in[17];
  const float* sw2re = (const float*)d_in[18]; const float* sw2im = (const float*)d_in[19];

  float* out = (float*)d_out;
  float* zout = out + (size_t)M_ * B_ * C_ * HW;   // z output, offset 33,554,432
  float* ws = (float*)d_ws;

  // 0) pack reused B panels (weights + twiddles) into L2-resident fragment pool
  k_pack<<<22, 256, 0, stream>>>(c_w, wq_w, wk_w, wa_w, w_w, wz_w, ws);
  // 1) attention aggregation -> z
  k_atn<<<NPTS / 16, 32, 0, stream>>>(x0, x1, x2, x3, c_b, wq_b, wk_b, wa_b,
                                      ws, zout);
  // 2..6) spectral path (partial-mode DFT GEMMs)
  k_dft_w<<<2048, 256, 0, stream>>>(x0, x1, x2, x3, ws);
  k_dft_h<<<256, 256, 0, stream>>>(ws);
  k_mix<<<256, 256, 0, stream>>>(sw1re, sw1im, sw2re, sw2im, ws);
  k_idft_h<<<256, 256, 0, stream>>>(ws);
  k_idft_w<<<2048, 256, 0, stream>>>(ws);
  // 7) pointwise convs + spectral add + GELU
  k_out<<<4096, 256, 0, stream>>>(x0, x1, x2, x3, w_b, wz_b, zout, ws, out);
}
